// GraphConstructor_7232724927020
// MI455X (gfx1250) — matrix operand order repaired and checked by reading, not compiled, not run on hardware
//
#include <hip/hip_runtime.h>
#include <hip/hip_bf16.h>

// ---------------------------------------------------------------------------
// Pearson-correlation graph constructor for MI455X (gfx1250, wave32, WMMA).
//
// Pass 1: z-normalize each [512]-series (ddof=1), emit f16 into d_ws (64 MB).
// Pass 2: per (batch, 32-row block) workgroup: compute the 32x1024 corr stripe
//         with v_wmma_f32_16x16x32_f16 in 8 phases of 128 columns.
//         2-row-group register blocking: one B fragment feeds two WMMAs.
//         32 rows/WG halves L2 B-traffic vs 16 rows/WG (2 GB total);
//         barriers inside the phase loop keep A frags in LDS (low VGPR count,
//         good occupancy). Fused masked top-5 per row, direct edge emission.
// ---------------------------------------------------------------------------

typedef __attribute__((ext_vector_type(16))) _Float16 v16h;
typedef __attribute__((ext_vector_type(8)))  _Float16 v8h;
typedef __attribute__((ext_vector_type(8)))  float    v8f;
typedef __attribute__((ext_vector_type(4)))  float    v4f;

#define BATCH 64
#define NVAR  1024
#define SLEN  512
#define TOPK  5

#define ROWS   32                 // rows of corr per workgroup
#define PHCOLS 128                // columns per phase (8 phases)
#define A_PAD  8                  // halves: row stride 520h=1040B -> 4 banks/row
#define C_PAD  4                  // floats: row stride 132f -> 4 banks/row

// ---------------------------------------------------------------------------
// Kernel 1: per-row normalization. One wave32 per row of 512 floats.
// ---------------------------------------------------------------------------
__global__ __launch_bounds__(256) void pearson_normalize(
    const float* __restrict__ x, _Float16* __restrict__ xh) {
  const int row  = blockIdx.x * 8 + (threadIdx.x >> 5);   // 8 waves / block
  const int lane = threadIdx.x & 31;

  const float* p = x + (size_t)row * SLEN + lane * 16;
  float v[16];
  {
    v4f a0 = *(const v4f*)(p);
    v4f a1 = *(const v4f*)(p + 4);
    v4f a2 = *(const v4f*)(p + 8);
    v4f a3 = *(const v4f*)(p + 12);
#pragma unroll
    for (int i = 0; i < 4; ++i) {
      v[i]      = a0[i];
      v[4 + i]  = a1[i];
      v[8 + i]  = a2[i];
      v[12 + i] = a3[i];
    }
  }

  float sum = 0.f, ssq = 0.f;
#pragma unroll
  for (int i = 0; i < 16; ++i) { sum += v[i]; ssq += v[i] * v[i]; }
#pragma unroll
  for (int off = 16; off > 0; off >>= 1) {
    sum += __shfl_xor(sum, off, 32);
    ssq += __shfl_xor(ssq, off, 32);
  }

  const float mean = sum * (1.0f / SLEN);
  float var = (ssq - (float)SLEN * mean * mean) * (1.0f / (SLEN - 1));
  var = var > 0.f ? var : 0.f;
  const float inv = 1.0f / (__builtin_sqrtf(var) + 1e-8f);

  v16h h;
#pragma unroll
  for (int i = 0; i < 16; ++i) h[i] = (_Float16)((v[i] - mean) * inv);
  *(v16h*)(xh + (size_t)row * SLEN + lane * 16) = h;
}

// ---------------------------------------------------------------------------
// Kernel 2: fused corr GEMM (WMMA f16 -> f32) + masked top-5 + edge emit.
// grid = (row blocks = 32, batches = 64), block = 256 (8 waves).
// ---------------------------------------------------------------------------
__global__ __launch_bounds__(256) void pearson_corr_topk(
    const _Float16* __restrict__ xh,
    long long* __restrict__ eidx, float* __restrict__ ew) {
  __shared__ _Float16 Alds[ROWS][SLEN + A_PAD];      // 33.3 KB
  __shared__ float    Clds[ROWS][PHCOLS + C_PAD];    // 16.9 KB (total ~50 KB)

  const int b    = blockIdx.y;
  const int row0 = blockIdx.x * ROWS;
  const int tid  = threadIdx.x;
  const int wave = tid >> 5;
  const int lane = tid & 31;
  const int l15  = lane & 15;
  const int hsel = lane >> 4;                        // lane half: 0 / 1

  // ---- stage A stripe: rows row0..row0+31, 512 halves each, into LDS ----
  {
    const int r  = tid >> 3;                         // 0..31
    const int c0 = (tid & 7) * 64;                   // 0..448
    const _Float16* g = xh + ((size_t)b * NVAR + row0 + r) * SLEN + c0;
#pragma unroll
    for (int q = 0; q < 8; ++q)
      *(v8h*)&Alds[r][c0 + 8 * q] = *(const v8h*)(g + 8 * q);
  }
  __syncthreads();

  // per-row top-5: each wave owns 4 rows (wave*4 .. wave*4+3)
  float tv[4][TOPK];
  int   ti[4][TOPK];
#pragma unroll
  for (int rr = 0; rr < 4; ++rr)
#pragma unroll
    for (int k = 0; k < TOPK; ++k) { tv[rr][k] = -1e30f; ti[rr][k] = -1; }

  const float inv_s = 1.0f / (float)SLEN;

#pragma unroll 1
  for (int phase = 0; phase < NVAR / PHCOLS; ++phase) {
    const int colbase = phase * PHCOLS;
    const int nloc    = wave * 16;                   // this wave's column tile

    // B fragment source: column (colbase+nloc+l15) of corr == that row of xn.
    // lanes 0-15 supply K=kk..kk+15, lanes 16-31 supply K=kk+16..kk+31.
    const _Float16* brow =
        xh + ((size_t)b * NVAR + colbase + nloc + l15) * SLEN + hsel * 16;
    __builtin_prefetch(brow, 0, 3);                  // global_prefetch_b8

    v8f acc0 = {};                                   // rows row0 .. row0+15
    v8f acc1 = {};                                   // rows row0+16 .. row0+31
#pragma unroll
    for (int kk = 0; kk < SLEN; kk += 32) {
      const int k0 = kk + hsel * 16;
      v16h bfrag = *(const v16h*)(brow + kk);

      // A fragments (contiguous, matches B): lanes 0-15 hold K kk..kk+15,
      // lanes 16-31 hold K kk+16..kk+31; row = l15 (+16 for grp 1).
      v8h a0lo = *(const v8h*)&Alds[l15][k0];
      v8h a0hi = *(const v8h*)&Alds[l15][k0 + 8];
      v8h a1lo = *(const v8h*)&Alds[16 + l15][k0];
      v8h a1hi = *(const v8h*)&Alds[16 + l15][k0 + 8];
      v16h af0, af1;
#pragma unroll
      for (int e = 0; e < 8; ++e) {
        af0[e] = a0lo[e]; af0[e + 8] = a0hi[e];
        af1[e] = a1lo[e]; af1[e + 8] = a1hi[e];
      }

      acc0 = __builtin_amdgcn_wmma_f32_16x16x32_f16(
          false, af0, false, bfrag, (short)0, acc0, false, false);
      acc1 = __builtin_amdgcn_wmma_f32_16x16x32_f16(
          false, af1, false, bfrag, (short)0, acc1, false, false);
    }

    // C layout: lanes 0-15 -> tile rows 0..7, lanes 16-31 -> 8..15, col = l15
    const int crow = hsel * 8;
#pragma unroll
    for (int j = 0; j < 8; ++j) {
      Clds[crow + j][nloc + l15]      = acc0[j] * inv_s;
      Clds[16 + crow + j][nloc + l15] = acc1[j] * inv_s;
    }
    __syncthreads();

    // ---- partial top-5 over this phase's 128 columns (4 rows / wave) ----
#pragma unroll
    for (int rr = 0; rr < 4; ++rr) {
      const int r       = wave * 4 + rr;
      const int selfcol = row0 + r;                  // diagonal mask target
#pragma unroll
      for (int step = 0; step < PHCOLS / 32; ++step) {
        const int cl = step * 32 + lane;             // conflict-free LDS scan
        const int c  = colbase + cl;
        float vv = Clds[r][cl];
        if (c == selfcol) vv = -1e30f;
        int ci = c;
#pragma unroll
        for (int k = 0; k < TOPK; ++k) {             // sorted bubble-insert
          if (vv > tv[rr][k]) {
            float tf = tv[rr][k]; tv[rr][k] = vv; vv = tf;
            int   tt = ti[rr][k]; ti[rr][k] = ci; ci = tt;
          }
        }
      }
    }
    __syncthreads();                                 // before Clds reuse
  }

  // ---- cross-lane merge of 32 sorted 5-lists -> global top-5, emit edges ----
  const long long E = (long long)BATCH * NVAR * TOPK;
#pragma unroll
  for (int rr = 0; rr < 4; ++rr) {
    const int r = wave * 4 + rr;
    int head = 0;
#pragma unroll
    for (int k = 0; k < TOPK; ++k) {
      float cv = (head < TOPK) ? tv[rr][head] : -1e30f;
      int   ci = (head < TOPK) ? ti[rr][head] : -1;
#pragma unroll
      for (int off = 16; off > 0; off >>= 1) {       // wave32 butterfly arg-max
        float ov = __shfl_xor(cv, off, 32);
        int   oi = __shfl_xor(ci, off, 32);
        if (ov > cv || (ov == cv && oi > ci)) { cv = ov; ci = oi; }
      }
      if (head < TOPK && ti[rr][head] == ci) head++; // unique winner pops
      if (lane == 0) {
        const long long src = (long long)b * NVAR + row0 + r;
        const long long dst = (long long)b * NVAR + ci;
        const long long e   = src * TOPK + k;
        eidx[e]     = src;
        eidx[E + e] = dst;
        ew[e]       = cv;
      }
    }
  }
}

// ---------------------------------------------------------------------------
extern "C" void kernel_launch(void* const* d_in, const int* in_sizes, int n_in,
                              void* d_out, int out_size, void* d_ws, size_t ws_size,
                              hipStream_t stream) {
  (void)in_sizes; (void)n_in; (void)out_size; (void)ws_size;

  const float* x  = (const float*)d_in[0];
  _Float16*    xh = (_Float16*)d_ws;            // needs 64*1024*512*2 = 64 MB

  // d_out layout: [edge_index int64: 2*E] then [edge_weight f32: E]
  const long long E   = (long long)BATCH * NVAR * TOPK;
  long long* eidx     = (long long*)d_out;
  float*     eweight  = (float*)(eidx + 2 * E);

  pearson_normalize<<<dim3((BATCH * NVAR) / 8), 256, 0, stream>>>(x, xh);
  pearson_corr_topk<<<dim3(NVAR / ROWS, BATCH), 256, 0, stream>>>(xh, eidx, eweight);
}
